// MSTParserLSTMModel_37615323578879
// MI455X (gfx1250) — matrix operand order repaired and checked by reading, not compiled, not run on hardware
//
#include <hip/hip_runtime.h>
#include <hip/hip_bf16.h>

typedef __attribute__((ext_vector_type(16))) __bf16 v16bf;
typedef __attribute__((ext_vector_type(8)))  float  v8f;

#define LH   125   // lstm hidden
#define G4   500   // 4*L gate width
#define GP   512   // padded gate width
#define HP   256   // padded 2L (250 -> 256)
#define XP   128   // padded hidden/embedding dim (125 -> 128)
#define HD   100   // arc-scorer hidden
#define HDP  128   // padded arc hidden

// gfx1250 has V_TANH_F32 (hardware tanh in the trans pipe)
#if __has_builtin(__builtin_amdgcn_tanhf)
__device__ __forceinline__ float fast_tanh(float x) { return __builtin_amdgcn_tanhf(x); }
#else
__device__ __forceinline__ float fast_tanh(float x) { return tanhf(x); }
#endif
// sigmoid(x) = 0.5*tanh(0.5x)+0.5 : one trans op instead of exp+rcp
__device__ __forceinline__ float fast_sig(float x) { return 0.5f * fast_tanh(0.5f * x) + 0.5f; }

// ---------------- prep kernels ----------------

// x_bf16[T][XP]: cols 0..99 = W_emb[word], 100..124 = P_emb[pos], rest 0
__global__ void embed_kernel(const int* __restrict__ wid, const int* __restrict__ pid,
                             const float* __restrict__ Wemb, const float* __restrict__ Pemb,
                             __bf16* __restrict__ x, int T) {
  int i = blockIdx.x * blockDim.x + threadIdx.x;
  if (i >= T * XP) return;
  int t = i / XP, k = i % XP;
  float v = 0.0f;
  if (k < 100)      v = Wemb[wid[t] * 100 + k];
  else if (k < 125) v = Pemb[pid[t] * 25 + (k - 100)];
  x[i] = (__bf16)v;
}

// pack f32 [N][K] row-major -> bf16 [Npad][Kpad], zero padded
__global__ void pack_w_kernel(const float* __restrict__ W, __bf16* __restrict__ out,
                              int N, int K, int Npad, int Kpad) {
  int i = blockIdx.x * blockDim.x + threadIdx.x;
  if (i >= Npad * Kpad) return;
  int n = i / Kpad, k = i % Kpad;
  out[i] = (__bf16)((n < N && k < K) ? W[n * K + k] : 0.0f);
}

// transpose-pack: in is [K][N] row-major (e.g. hidLayerFOH [250][100]) -> bf16 [Npad][Kpad]
__global__ void pack_wt_kernel(const float* __restrict__ W, __bf16* __restrict__ out,
                               int N, int K, int Npad, int Kpad) {
  int i = blockIdx.x * blockDim.x + threadIdx.x;
  if (i >= Npad * Kpad) return;
  int n = i / Kpad, k = i % Kpad;
  out[i] = (__bf16)((n < N && k < K) ? W[k * N + n] : 0.0f);
}

__global__ void zero_bf16_kernel(__bf16* __restrict__ p, int n) {
  int i = blockIdx.x * blockDim.x + threadIdx.x;
  if (i < n) p[i] = (__bf16)0.0f;
}

// ---------------- WMMA GEMM (bf16 NT) ----------------
// C[M][Npad] = A[M][KPAD] * B[Npad][KPAD]^T (+bias on cols < nBias).
// One wave computes a 16(M) x 64(N) tile; KPAD compile-time -> fully unrolled k loop.
template <int KPAD>
__global__ void gemm_bf16_nt(const __bf16* __restrict__ A, const __bf16* __restrict__ B,
                             float* __restrict__ C, const float* __restrict__ bias, int nBias,
                             int Npad) {
  const int tilesN = Npad >> 6;
  const int wid    = blockIdx.x * (blockDim.x >> 5) + (threadIdx.x >> 5);
  const int mTile  = wid / tilesN;
  const int nTile  = wid % tilesN;
  const int m0     = mTile << 4;
  const int n0     = nTile << 6;
  const int lane   = threadIdx.x & 31;
  const int half   = lane >> 4;
  const int lm     = lane & 15;

  v8f acc0 = {}, acc1 = {}, acc2 = {}, acc3 = {};

  #pragma unroll
  for (int k0 = 0; k0 < KPAD; k0 += 32) {
    // A fragment: row m0+lm; elem e -> K = k0 + 16*half + e
    v16bf a;
    const __bf16* arow = A + (size_t)(m0 + lm) * KPAD + k0 + 16 * half;
    #pragma unroll
    for (int e = 0; e < 8; ++e) a[e] = arow[e];
    #pragma unroll
    for (int e = 0; e < 8; ++e) a[8 + e] = arow[8 + e];

    // B fragments: col n0+s*16+lm; elem e -> K = k0 + 16*half + e
    #pragma unroll
    for (int s = 0; s < 4; ++s) {
      const __bf16* brow = B + (size_t)(n0 + s * 16 + lm) * KPAD + k0 + 16 * half;
      v16bf b;
      #pragma unroll
      for (int e = 0; e < 16; ++e) b[e] = brow[e];
      v8f* acc = (s == 0) ? &acc0 : (s == 1) ? &acc1 : (s == 2) ? &acc2 : &acc3;
      *acc = __builtin_amdgcn_wmma_f32_16x16x32_bf16(false, a, false, b,
                                                     (short)0, *acc, false, false);
    }
  }

  #pragma unroll
  for (int s = 0; s < 4; ++s) {
    const v8f& acc = (s == 0) ? acc0 : (s == 1) ? acc1 : (s == 2) ? acc2 : acc3;
    int col = n0 + s * 16 + lm;
    float bv = (bias != nullptr && col < nBias) ? bias[col] : 0.0f;
    #pragma unroll
    for (int r = 0; r < 8; ++r) {
      C[(size_t)(m0 + r + 8 * half) * Npad + col] = acc[r] + bv;
    }
  }
}

// ---------------- recurrent LSTM layer: WMMA matvec, weights resident in VGPRs ----
// One workgroup (16 waves / 512 threads) per direction; blockIdx 0 = fwd, 1 = bwd.
// Each wave owns 2 gate tiles (16 rows each): 2x4x8 = 64 weight VGPRs/wave, no spills.
// Whh packed bf16 [GP][XP]. Gpre[t][GP] holds x@Wih^T + b.
// Writes bf16 h into Hout[t][HP]: bwd -> cols 0..124, fwd -> cols 125..249.
__global__ __launch_bounds__(512, 1)
void lstm_layer_wmma_kernel(const float* __restrict__ GpreF,
                            const float* __restrict__ GpreB,
                            const __bf16* __restrict__ WhhFb,
                            const __bf16* __restrict__ WhhBb,
                            __bf16* __restrict__ Hout, int T) {
  const float* Gpre;
  const __bf16* Whh;
  int rev, colOff;
  if (blockIdx.x == 0) { Gpre = GpreF; Whh = WhhFb; rev = 0; colOff = LH; }
  else                 { Gpre = GpreB; Whh = WhhBb; rev = 1; colOff = 0;  }

  __shared__ alignas(32) __bf16 hbf[XP];   // recurrent state, bf16
  __shared__ alignas(32) float  g[GP];     // wmma matvec output

  const int tid  = threadIdx.x;            // 512
  const int wave = tid >> 5;               // 0..15 -> gate tiles wave*2, wave*2+1
  const int lane = tid & 31;
  const int half = lane >> 4;
  const int lm   = lane & 15;

  // Preload Whh A-fragments into registers once (resident for all T steps).
  v16bf afrag[2][4];                       // [tile][kstep]
  #pragma unroll
  for (int tle = 0; tle < 2; ++tle) {
    const __bf16* wrow = Whh + (size_t)((wave * 2 + tle) * 16 + lm) * XP;
    #pragma unroll
    for (int ks = 0; ks < 4; ++ks) {
      const __bf16* ar = wrow + ks * 32 + 16 * half;
      #pragma unroll
      for (int e = 0; e < 8; ++e) afrag[tle][ks][e] = ar[e];
      #pragma unroll
      for (int e = 0; e < 8; ++e) afrag[tle][ks][8 + e] = ar[8 + e];
    }
  }

  if (tid < XP) hbf[tid] = (__bf16)0.0f;
  float creg = 0.0f;

  // prefetch first step's gate pre-activations
  float p0 = 0.f, p1 = 0.f, p2 = 0.f, p3 = 0.f;
  if (tid < LH) {
    const float* gp = Gpre + (size_t)(rev ? T - 1 : 0) * GP;
    p0 = gp[tid];
    p1 = gp[LH + tid];
    p2 = gp[2 * LH + tid];
    p3 = gp[3 * LH + tid];
  }
  __syncthreads();

  for (int s = 0; s < T; ++s) {
    const int t = rev ? (T - 1 - s) : s;

    // B fragments: h broadcast across all 16 columns; elem e -> k = ks*32 + 16*half + e
    v16bf bfrag[4];
    #pragma unroll
    for (int ks = 0; ks < 4; ++ks) {
      bfrag[ks] = *reinterpret_cast<const v16bf*>(&hbf[ks * 32 + 16 * half]);
    }

    #pragma unroll
    for (int tle = 0; tle < 2; ++tle) {
      v8f acc = {};
      #pragma unroll
      for (int ks = 0; ks < 4; ++ks) {
        acc = __builtin_amdgcn_wmma_f32_16x16x32_bf16(false, afrag[tle][ks], false, bfrag[ks],
                                                      (short)0, acc, false, false);
      }
      // column 0 of D: lane 0 -> rows tile*16+0..7, lane 16 -> rows tile*16+8..15
      if (lm == 0) {
        *reinterpret_cast<v8f*>(&g[(wave * 2 + tle) * 16 + 8 * half]) = acc;
      }
    }
    __syncthreads();

    if (tid < LH) {
      float ig = fast_sig (g[tid]          + p0);
      float fg = fast_sig (g[LH + tid]     + p1);
      float gg = fast_tanh(g[2 * LH + tid] + p2);
      float og = fast_sig (g[3 * LH + tid] + p3);
      creg = fg * creg + ig * gg;
      float hn = og * fast_tanh(creg);
      hbf[tid] = (__bf16)hn;
      Hout[(size_t)t * HP + colOff + tid] = (__bf16)hn;
      // prefetch next step's Gpre so the L2 latency overlaps the next WMMA phase
      if (s + 1 < T) {
        const float* gp = Gpre + (size_t)(rev ? (T - 2 - s) : (s + 1)) * GP;
        p0 = gp[tid];
        p1 = gp[LH + tid];
        p2 = gp[2 * LH + tid];
        p3 = gp[3 * LH + tid];
      }
    }
    __syncthreads();
  }
}

// ---------------- pair scorer ----------------
// scores[i][j] = sum_k tanh(head[i][k] + mod[j][k]) * w[k]   (hidBias folded into head)
// 32x32 tile per block, 2x2 register micro-tile per thread.
__global__ void pair_kernel(const float* __restrict__ head, const float* __restrict__ mod,
                            const float* __restrict__ outW, float* __restrict__ scores, int T) {
  __shared__ float sh[32][HD + 4];
  __shared__ float sm[32][HD + 4];
  __shared__ float sw[HD];

  const int bi  = blockIdx.y * 32;
  const int bj  = blockIdx.x * 32;
  const int tid = threadIdx.x;             // 256 threads

  for (int idx = tid; idx < 32 * HD; idx += 256) {
    int r = idx / HD, k = idx % HD;
    sh[r][k] = head[(size_t)(bi + r) * HDP + k];
    sm[r][k] = mod[(size_t)(bj + r) * HDP + k];
  }
  if (tid < HD) sw[tid] = outW[tid];
  __syncthreads();

  const int r0 = tid >> 4, r1 = r0 + 16;
  const int c0 = tid & 15, c1 = c0 + 16;
  float a00 = 0.f, a01 = 0.f, a10 = 0.f, a11 = 0.f;
  #pragma unroll 4
  for (int k = 0; k < HD; ++k) {
    float w  = sw[k];
    float h0 = sh[r0][k], h1 = sh[r1][k];
    float m0 = sm[c0][k], m1 = sm[c1][k];
    a00 += fast_tanh(h0 + m0) * w;
    a01 += fast_tanh(h0 + m1) * w;
    a10 += fast_tanh(h1 + m0) * w;
    a11 += fast_tanh(h1 + m1) * w;
  }
  scores[(size_t)(bi + r0) * T + bj + c0] = a00;
  scores[(size_t)(bi + r0) * T + bj + c1] = a01;
  scores[(size_t)(bi + r1) * T + bj + c0] = a10;
  scores[(size_t)(bi + r1) * T + bj + c1] = a11;
}

// ---------------- host side ----------------

static inline size_t align256(size_t x) { return (x + 255) & ~(size_t)255; }

static void launch_gemm(const __bf16* A, const __bf16* B, float* C,
                        const float* bias, int nBias,
                        int M, int Npad, int Kpad, hipStream_t s) {
  int waves  = (M / 16) * (Npad / 64);
  int blocks = waves / 8;                  // 8 waves (256 threads) per block
  if (Kpad == 128)
    gemm_bf16_nt<128><<<blocks, 256, 0, s>>>(A, B, C, bias, nBias, Npad);
  else
    gemm_bf16_nt<256><<<blocks, 256, 0, s>>>(A, B, C, bias, nBias, Npad);
}

extern "C" void kernel_launch(void* const* d_in, const int* in_sizes, int n_in,
                              void* d_out, int out_size, void* d_ws, size_t ws_size,
                              hipStream_t stream) {
  const int*   word_ids = (const int*)  d_in[0];
  const int*   pos_ids  = (const int*)  d_in[1];
  const float* W_emb    = (const float*)d_in[2];
  const float* P_emb    = (const float*)d_in[3];
  const float* Wih_f0   = (const float*)d_in[4];
  const float* Whh_f0   = (const float*)d_in[5];
  const float* b_f0     = (const float*)d_in[6];
  const float* Wih_b0   = (const float*)d_in[7];
  const float* Whh_b0   = (const float*)d_in[8];
  const float* b_b0     = (const float*)d_in[9];
  const float* Wih_f1   = (const float*)d_in[10];
  const float* Whh_f1   = (const float*)d_in[11];
  const float* b_f1     = (const float*)d_in[12];
  const float* Wih_b1   = (const float*)d_in[13];
  const float* Whh_b1   = (const float*)d_in[14];
  const float* b_b1     = (const float*)d_in[15];
  const float* FOH      = (const float*)d_in[16];
  const float* FOM      = (const float*)d_in[17];
  const float* hidBias  = (const float*)d_in[18];
  const float* outW     = (const float*)d_in[19];
  float* scores = (float*)d_out;

  const int T = in_sizes[0];               // 1024

  // workspace carve-up
  char* p = (char*)d_ws;
  size_t off = 0;
  auto carve = [&](size_t bytes) { void* r = p + off; off += align256(bytes); return r; };

  __bf16* x_bf   = (__bf16*)carve((size_t)T * XP * 2);
  __bf16* Wf0    = (__bf16*)carve((size_t)GP * XP * 2);
  __bf16* Wb0    = (__bf16*)carve((size_t)GP * XP * 2);
  __bf16* Wf1    = (__bf16*)carve((size_t)GP * HP * 2);
  __bf16* Wb1    = (__bf16*)carve((size_t)GP * HP * 2);
  __bf16* Rf0    = (__bf16*)carve((size_t)GP * XP * 2);   // Whh packed bf16
  __bf16* Rb0    = (__bf16*)carve((size_t)GP * XP * 2);
  __bf16* Rf1    = (__bf16*)carve((size_t)GP * XP * 2);
  __bf16* Rb1    = (__bf16*)carve((size_t)GP * XP * 2);
  __bf16* FOHt   = (__bf16*)carve((size_t)HDP * HP * 2);
  __bf16* FOMt   = (__bf16*)carve((size_t)HDP * HP * 2);
  float*  Gf0    = (float*) carve((size_t)T * GP * 4);
  float*  Gb0    = (float*) carve((size_t)T * GP * 4);
  float*  Gf1    = (float*) carve((size_t)T * GP * 4);
  float*  Gb1    = (float*) carve((size_t)T * GP * 4);
  __bf16* h1_bf  = (__bf16*)carve((size_t)T * HP * 2);
  __bf16* h2_bf  = (__bf16*)carve((size_t)T * HP * 2);
  float*  headB  = (float*) carve((size_t)T * HDP * 4);
  float*  modB   = (float*) carve((size_t)T * HDP * 4);
  (void)ws_size; (void)n_in; (void)out_size;

  const int BT = 256;
  auto blocksFor = [&](int n) { return (n + BT - 1) / BT; };

  // ---- stage 0: packing / embedding ----
  embed_kernel<<<blocksFor(T * XP), BT, 0, stream>>>(word_ids, pos_ids, W_emb, P_emb, x_bf, T);
  pack_w_kernel<<<blocksFor(GP * XP), BT, 0, stream>>>(Wih_f0, Wf0, G4, LH, GP, XP);
  pack_w_kernel<<<blocksFor(GP * XP), BT, 0, stream>>>(Wih_b0, Wb0, G4, LH, GP, XP);
  pack_w_kernel<<<blocksFor(GP * HP), BT, 0, stream>>>(Wih_f1, Wf1, G4, 2 * LH, GP, HP);
  pack_w_kernel<<<blocksFor(GP * HP), BT, 0, stream>>>(Wih_b1, Wb1, G4, 2 * LH, GP, HP);
  pack_w_kernel<<<blocksFor(GP * XP), BT, 0, stream>>>(Whh_f0, Rf0, G4, LH, GP, XP);
  pack_w_kernel<<<blocksFor(GP * XP), BT, 0, stream>>>(Whh_b0, Rb0, G4, LH, GP, XP);
  pack_w_kernel<<<blocksFor(GP * XP), BT, 0, stream>>>(Whh_f1, Rf1, G4, LH, GP, XP);
  pack_w_kernel<<<blocksFor(GP * XP), BT, 0, stream>>>(Whh_b1, Rb1, G4, LH, GP, XP);
  pack_wt_kernel<<<blocksFor(HDP * HP), BT, 0, stream>>>(FOH, FOHt, HD, 2 * LH, HDP, HP);
  pack_wt_kernel<<<blocksFor(HDP * HP), BT, 0, stream>>>(FOM, FOMt, HD, 2 * LH, HDP, HP);
  zero_bf16_kernel<<<blocksFor(T * HP), BT, 0, stream>>>(h1_bf, T * HP);
  zero_bf16_kernel<<<blocksFor(T * HP), BT, 0, stream>>>(h2_bf, T * HP);

  // ---- stage 1: layer-0 input projections (WMMA) + WMMA recurrence ----
  launch_gemm(x_bf, Wf0, Gf0, b_f0, G4, T, GP, XP, stream);
  launch_gemm(x_bf, Wb0, Gb0, b_b0, G4, T, GP, XP, stream);
  lstm_layer_wmma_kernel<<<2, 512, 0, stream>>>(Gf0, Gb0, Rf0, Rb0, h1_bf, T);

  // ---- stage 2: layer-1 input projections (WMMA) + WMMA recurrence ----
  launch_gemm(h1_bf, Wf1, Gf1, b_f1, G4, T, GP, HP, stream);
  launch_gemm(h1_bf, Wb1, Gb1, b_b1, G4, T, GP, HP, stream);
  lstm_layer_wmma_kernel<<<2, 512, 0, stream>>>(Gf1, Gb1, Rf1, Rb1, h2_bf, T);

  // ---- stage 3: head/mod projections (WMMA); fold hidBias into head ----
  launch_gemm(h2_bf, FOHt, headB, hidBias, HD, T, HDP, HP, stream);
  launch_gemm(h2_bf, FOMt, modB, nullptr, 0, T, HDP, HP, stream);

  // ---- stage 4: pairwise tanh scorer ----
  dim3 pgrid(T / 32, T / 32);
  pair_kernel<<<pgrid, 256, 0, stream>>>(headB, modB, outW, scores, T);
}
